// BohaoDecoder_57492432224789
// MI455X (gfx1250) — matrix operand order repaired and checked by reading, not compiled, not run on hardware
//
#include <hip/hip_runtime.h>
#include <hip/hip_bf16.h>
#include <math.h>

// ---------------- problem constants (match reference) ----------------
#define D_MODEL 1024
#define NHEADS  16
#define DHEAD   64
#define NLAYERS 2
#define SEQLEN  512
#define BATCH   64
#define TSTEPS  48
#define LN_EPS  1e-5f

typedef __bf16 bf16_t;
typedef __bf16 v16bf __attribute__((ext_vector_type(16)));
typedef float  v8f   __attribute__((ext_vector_type(8)));
typedef float  v4f   __attribute__((ext_vector_type(4)));

// low 32 bits of a flat shared pointer == wave-relative LDS byte offset
#define LDS_OFF(p) ((unsigned)(unsigned long long)(p))

// one 16-byte async global->LDS copy per lane (ASYNCcnt-tracked)
__device__ __forceinline__ void async_copy16(unsigned lds_off, const void* gptr) {
  asm volatile("global_load_async_to_lds_b128 %0, %1, off"
               :: "v"(lds_off), "v"(gptr) : "memory");
}

// ---------------- WMMA fragment builders (gfx1250 wave32 layouts) ----------------
// A fragment 16x32 bf16 from f32: lane = 16h+m holds K={kk+8h..+7, kk+16+8h..+7} of row m.
__device__ __forceinline__ v16bf a_frag_f32(const float* __restrict__ row,
                                            int kk, int h) {
  v4f a0 = *(const v4f*)(row + kk + h * 8);
  v4f a1 = *(const v4f*)(row + kk + h * 8 + 4);
  v4f b0 = *(const v4f*)(row + kk + 16 + h * 8);
  v4f b1 = *(const v4f*)(row + kk + 16 + h * 8 + 4);
  v16bf r;
#pragma unroll
  for (int i = 0; i < 4; ++i) {
    r[i]      = (__bf16)a0[i];
    r[4 + i]  = (__bf16)a1[i];
    r[8 + i]  = (__bf16)b0[i];
    r[12 + i] = (__bf16)b1[i];
  }
  return r;
}

// B fragment 32x16 bf16 (KxN), B[k,n]=W[n,k]: lane = 16h+n holds K={kk+8h..+7, kk+16+8h..+7} of W row n.
__device__ __forceinline__ v16bf b_frag_bf16(const bf16_t* __restrict__ row,
                                             int kk, int h) {
  union { v4f f[2]; v16bf bf; } u;
  u.f[0] = *(const v4f*)(row + kk + h * 8);
  u.f[1] = *(const v4f*)(row + kk + 16 + h * 8);
  return u.bf;
}

// ---------------- LDS-staged, async double-buffered skinny GEMM ----------------
// out[m, n] = act(A[64,K] @ W[N,K]^T + bias [+ res]);  one block per 64-column panel.
// Block = 256 thr (8 waves); wave owns 1 M-tile x 2 N-tiles (A-frag feeds 2 WMMAs).
// Per 64-wide K-chunk each wave issues 6 async b128 copies; s_wait_asynccnt 6 keeps
// the next chunk in flight while the current one is consumed from LDS.
__device__ __forceinline__ void issue_chunk(const float* __restrict__ A,
                                            const bf16_t* __restrict__ W, int K,
                                            int nblk, int kc,
                                            float* As, bf16_t* Bs, int tid) {
#pragma unroll
  for (int i = 0; i < 4; ++i) {                     // A chunk: 64x64 f32 = 16KB
    int row = i * 16 + (tid >> 4);
    int col = (tid & 15) * 4;
    async_copy16(LDS_OFF(As + row * 64 + col),
                 A + (size_t)row * K + kc * 64 + col);
  }
#pragma unroll
  for (int i = 0; i < 2; ++i) {                     // B chunk: 64x64 bf16 = 8KB
    int row = i * 32 + (tid >> 3);
    int col = (tid & 7) * 8;
    async_copy16(LDS_OFF(Bs + row * 64 + col),
                 W + (size_t)(nblk * 64 + row) * K + kc * 64 + col);
  }
}

__global__ __launch_bounds__(256) void wmma_gemm_staged_kernel(
    const float* __restrict__ A, const bf16_t* __restrict__ W,
    const float* __restrict__ bias, const float* __restrict__ res,
    float* __restrict__ out, long out_stride, int N, int K, int relu) {
  __shared__ float  As[2][64 * 64];                 // 32KB
  __shared__ bf16_t Bs[2][64 * 64];                 // 16KB
  int tid  = threadIdx.x;
  int w    = tid >> 5, lane = tid & 31;
  int h    = lane >> 4, r = lane & 15;
  int nblk = blockIdx.x;
  int mt   = w & 3;                                 // M-tile of this wave
  int ntb  = (w >> 2) * 2;                          // first of 2 local N-tiles
  int nchunk = K >> 6;

  issue_chunk(A, W, K, nblk, 0, As[0], Bs[0], tid);
  v8f acc[2] = {{0.f,0.f,0.f,0.f,0.f,0.f,0.f,0.f},
                {0.f,0.f,0.f,0.f,0.f,0.f,0.f,0.f}};
  for (int kc = 0; kc < nchunk; ++kc) {
    int cur = kc & 1;
    if (kc + 1 < nchunk) {
      issue_chunk(A, W, K, nblk, kc + 1, As[cur ^ 1], Bs[cur ^ 1], tid);
      asm volatile("s_wait_asynccnt 6" ::: "memory");   // chunk kc landed; kc+1 in flight
    } else {
      asm volatile("s_wait_asynccnt 0" ::: "memory");
    }
    __syncthreads();
    const float*  Ac = As[cur];
    const bf16_t* Bc = Bs[cur];
#pragma unroll
    for (int kk = 0; kk < 64; kk += 32) {
      v16bf af = a_frag_f32(Ac + (mt * 16 + r) * 64, kk, h);
      v16bf b0 = b_frag_bf16(Bc + ((ntb + 0) * 16 + r) * 64, kk, h);
      v16bf b1 = b_frag_bf16(Bc + ((ntb + 1) * 16 + r) * 64, kk, h);
      acc[0] = __builtin_amdgcn_wmma_f32_16x16x32_bf16(false, af, false, b0,
                                                       (short)0, acc[0], false, false);
      acc[1] = __builtin_amdgcn_wmma_f32_16x16x32_bf16(false, af, false, b1,
                                                       (short)0, acc[1], false, false);
    }
    __syncthreads();                                // buffer reusable for kc+2's issue
  }
#pragma unroll
  for (int tile = 0; tile < 2; ++tile) {
    int   n  = nblk * 64 + (ntb + tile) * 16 + r;   // C/D layout: n = lane&15
    float bv = bias ? bias[n] : 0.f;
#pragma unroll
    for (int v = 0; v < 8; ++v) {
      int m = mt * 16 + h * 8 + v;                  // C/D layout: m = 8h + vgpr
      float val = acc[tile][v] + bv;
      if (res)  val += res[(size_t)m * N + n];
      if (relu) val = fmaxf(val, 0.f);
      out[(size_t)m * out_stride + n] = val;
    }
  }
}

// ---------------- K/V precompute GEMM: memory[B*S,D] @ W^T -> bf16 [B,NH,S,DH] ----------------
// Streaming-A big GEMM; B (2MB) is L2-resident, A read once -> direct-global WMMA.
__global__ __launch_bounds__(256) void kv_gemm_kernel(
    const float* __restrict__ A, const bf16_t* __restrict__ W,
    const float* __restrict__ bias, bf16_t* __restrict__ dst) {
  const int K = D_MODEL;
  int wid  = (blockIdx.x * 256 + threadIdx.x) >> 5;
  int lane = threadIdx.x & 31;
  const int mtiles = (BATCH * SEQLEN) >> 4;         // 2048
  int mt = wid % mtiles, nt = wid / mtiles;
  int h = lane >> 4, r = lane & 15;
  const float*  arow = A + (size_t)(mt * 16 + r) * K;
  const bf16_t* wrow = W + (size_t)(nt * 16 + r) * K;
  v8f acc = {0.f, 0.f, 0.f, 0.f, 0.f, 0.f, 0.f, 0.f};
  for (int kb = 0; kb < K; kb += 32) {
    __builtin_prefetch(arow + kb + 128, 0, 1);      // global_prefetch_b8
    __builtin_prefetch(wrow + kb + 256, 0, 1);
    v16bf a = a_frag_f32(arow, kb, h);
    v16bf b = b_frag_bf16(wrow, kb, h);
    acc = __builtin_amdgcn_wmma_f32_16x16x32_bf16(false, a, false, b,
                                                  (short)0, acc, false, false);
  }
  int   n  = nt * 16 + r;
  int   hh = n >> 6, d = n & 63;
  float bv = bias[n];
#pragma unroll
  for (int v = 0; v < 8; ++v) {
    int m = mt * 16 + h * 8 + v;
    int b = m >> 9, s = m & 511;                    // m = b*512 + s
    dst[(((size_t)b * NHEADS + hh) * SEQLEN + s) * DHEAD + d] = (bf16_t)(acc[v] + bv);
  }
}

// ---------------- cross-attention (length-1 query -> batched GEMV, VALU) ----------------
// grid = B, block = 512 (16 waves, one wave per head)
__global__ __launch_bounds__(512) void attn_kernel(
    const float* __restrict__ q,      // [B, D]
    const bf16_t* __restrict__ memK,  // [B, NH, S, DH] (this layer)
    const bf16_t* __restrict__ memV,
    float* __restrict__ o) {          // [B, D]
  __shared__ float sc[NHEADS][SEQLEN];
  __shared__ float qs[NHEADS][DHEAD];
  int b    = blockIdx.x;
  int h    = threadIdx.x >> 5;
  int lane = threadIdx.x & 31;
  qs[h][lane * 2]     = q[b * D_MODEL + h * DHEAD + lane * 2];
  qs[h][lane * 2 + 1] = q[b * D_MODEL + h * DHEAD + lane * 2 + 1];
  const bf16_t* kbase = memK + ((size_t)b * NHEADS + h) * SEQLEN * DHEAD;
  for (int it = 0; it < 16; ++it) {
    int s = it * 32 + lane;
    const bf16_t* kr = kbase + (size_t)s * DHEAD;
    float acc = 0.f;
#pragma unroll
    for (int dc = 0; dc < 8; ++dc) {
      union { v4f f; __bf16 e[8]; } u;
      u.f = *(const v4f*)(kr + dc * 8);
#pragma unroll
      for (int j = 0; j < 8; ++j) acc += (float)u.e[j] * qs[h][dc * 8 + j];
    }
    sc[h][s] = acc * 0.125f;                        // 1/sqrt(64)
  }
  float mx = -1e30f;
  for (int it = 0; it < 16; ++it) mx = fmaxf(mx, sc[h][it * 32 + lane]);
  for (int off = 16; off; off >>= 1) mx = fmaxf(mx, __shfl_xor(mx, off, 32));
  float sum = 0.f;
  for (int it = 0; it < 16; ++it) {
    int s = it * 32 + lane;
    float e = __expf(sc[h][s] - mx);
    sc[h][s] = e;
    sum += e;
  }
  for (int off = 16; off; off >>= 1) sum += __shfl_xor(sum, off, 32);
  float inv = 1.f / sum;
  const bf16_t* vbase = memV + ((size_t)b * NHEADS + h) * SEQLEN * DHEAD;
  int   d0 = lane * 2;
  float o0 = 0.f, o1 = 0.f;
  for (int s = 0; s < SEQLEN; ++s) {
    float a = sc[h][s];
    union { float f; __bf16 e[2]; } u;
    u.f = *(const float*)(vbase + (size_t)s * DHEAD + d0);
    o0 += a * (float)u.e[0];
    o1 += a * (float)u.e[1];
  }
  o[b * D_MODEL + h * DHEAD + d0]     = o0 * inv;
  o[b * D_MODEL + h * DHEAD + d0 + 1] = o1 * inv;
}

// ---------------- LayerNorm over D=1024, one block per row ----------------
__global__ __launch_bounds__(256) void ln_kernel(
    const float* __restrict__ y, const float* __restrict__ g,
    const float* __restrict__ bb, float* __restrict__ xout) {
  __shared__ float rs[256], rq[256];
  int row = blockIdx.x, tid = threadIdx.x;
  const float* yr = y + (size_t)row * D_MODEL;
  v4f v = *(const v4f*)(yr + tid * 4);
  float s  = v[0] + v[1] + v[2] + v[3];
  float q2 = v[0] * v[0] + v[1] * v[1] + v[2] * v[2] + v[3] * v[3];
  rs[tid] = s; rq[tid] = q2;
  __syncthreads();
  for (int st = 128; st > 0; st >>= 1) {
    if (tid < st) { rs[tid] += rs[tid + st]; rq[tid] += rq[tid + st]; }
    __syncthreads();
  }
  float mean = rs[0] * (1.f / D_MODEL);
  float var  = rq[0] * (1.f / D_MODEL) - mean * mean;
  float rinv = rsqrtf(var + LN_EPS);
  v4f gv = *(const v4f*)(g + tid * 4);
  v4f bv = *(const v4f*)(bb + tid * 4);
  v4f o;
#pragma unroll
  for (int j = 0; j < 4; ++j) o[j] = (v[j] - mean) * rinv * gv[j] + bv[j];
  *(v4f*)(xout + (size_t)row * D_MODEL + tid * 4) = o;
}

// ---------------- gate: sigmoid(x . gate_w + b), one wave per batch row ----------------
__global__ __launch_bounds__(32) void gate_kernel(
    const float* __restrict__ x, const float* __restrict__ gw,
    const float* __restrict__ gb, float* __restrict__ out) {
  int b = blockIdx.x, lane = threadIdx.x;
  float s = 0.f;
  for (int i = lane; i < D_MODEL; i += 32) s += x[b * D_MODEL + i] * gw[i];
  for (int off = 16; off; off >>= 1) s += __shfl_xor(s, off, 32);
  if (lane == 0) out[b] = 1.f / (1.f + __expf(-(s + gb[0])));
}

// ---------------- helpers ----------------
__global__ __launch_bounds__(256) void f32_to_bf16_kernel(
    const float* __restrict__ src, bf16_t* __restrict__ dst, int n) {
  int i = (blockIdx.x * 256 + threadIdx.x) * 4;
  if (i + 3 < n) {
    v4f v = *(const v4f*)(src + i);
#pragma unroll
    for (int j = 0; j < 4; ++j) dst[i + j] = (bf16_t)v[j];
  }
}

__global__ __launch_bounds__(256) void zero_kernel(float* __restrict__ p, int n) {
  int i = blockIdx.x * 256 + threadIdx.x;
  if (i < n) p[i] = 0.f;
}

// ---------------- host orchestration ----------------
extern "C" void kernel_launch(void* const* d_in, const int* in_sizes, int n_in,
                              void* d_out, int out_size, void* d_ws, size_t ws_size,
                              hipStream_t stream) {
  (void)in_sizes; (void)n_in; (void)out_size; (void)ws_size;
  const float* memory = (const float*)d_in[0];
  // d_in[1] = memory_lengths: unused by the reference math.
  const float* sa_w  = (const float*)d_in[2];
  const float* sa_b  = (const float*)d_in[3];
  const float* sa_ow = (const float*)d_in[4];
  const float* sa_ob = (const float*)d_in[5];
  const float* ca_w  = (const float*)d_in[6];
  const float* ca_b  = (const float*)d_in[7];
  const float* ca_ow = (const float*)d_in[8];
  const float* ca_ob = (const float*)d_in[9];
  const float* ln1_g = (const float*)d_in[10];
  const float* ln1_b = (const float*)d_in[11];
  const float* ln2_g = (const float*)d_in[12];
  const float* ln2_b = (const float*)d_in[13];
  const float* ln3_g = (const float*)d_in[14];
  const float* ln3_b = (const float*)d_in[15];
  const float* ff1_w = (const float*)d_in[16];
  const float* ff1_b = (const float*)d_in[17];
  const float* ff2_w = (const float*)d_in[18];
  const float* ff2_b = (const float*)d_in[19];
  const float* glot_w = (const float*)d_in[20];
  const float* glot_b = (const float*)d_in[21];
  const float* gate_w = (const float*)d_in[22];
  const float* gate_b = (const float*)d_in[23];
  float* outf = (float*)d_out;

  const size_t DD   = (size_t)D_MODEL * D_MODEL;               // 1M elems
  const size_t KVSZ = (size_t)BATCH * NHEADS * SEQLEN * DHEAD; // per-layer K or V

  size_t off = 0;
  auto wsalloc = [&](size_t bytes) {
    void* p = (char*)d_ws + off;
    off += (bytes + 255) & ~(size_t)255;
    return p;
  };
  bf16_t *wsa_v[NLAYERS], *wsa_o[NLAYERS], *wca_q[NLAYERS], *wca_k[NLAYERS],
         *wca_v[NLAYERS], *wca_o[NLAYERS], *wff1[NLAYERS], *wff2[NLAYERS];
  for (int l = 0; l < NLAYERS; ++l) {
    wsa_v[l] = (bf16_t*)wsalloc(DD * 2); wsa_o[l] = (bf16_t*)wsalloc(DD * 2);
    wca_q[l] = (bf16_t*)wsalloc(DD * 2); wca_k[l] = (bf16_t*)wsalloc(DD * 2);
    wca_v[l] = (bf16_t*)wsalloc(DD * 2); wca_o[l] = (bf16_t*)wsalloc(DD * 2);
    wff1[l]  = (bf16_t*)wsalloc(DD * 2); wff2[l]  = (bf16_t*)wsalloc(DD * 2);
  }
  bf16_t* wglot = (bf16_t*)wsalloc(2 * DD * 2);
  bf16_t* memK  = (bf16_t*)wsalloc(NLAYERS * KVSZ * 2);
  bf16_t* memV  = (bf16_t*)wsalloc(NLAYERS * KVSZ * 2);
  float* x    = (float*)wsalloc(BATCH * D_MODEL * 4);
  float* ybuf = (float*)wsalloc(BATCH * D_MODEL * 4);
  float* vbuf = (float*)wsalloc(BATCH * D_MODEL * 4);
  float* qbuf = (float*)wsalloc(BATCH * D_MODEL * 4);
  float* obuf = (float*)wsalloc(BATCH * D_MODEL * 4);
  float* fbuf = (float*)wsalloc(BATCH * D_MODEL * 4);

  // ---- weight conversions to bf16 (all slices are contiguous) ----
  auto conv = [&](const float* s, bf16_t* d, size_t n) {
    f32_to_bf16_kernel<<<(unsigned)(n / 1024), 256, 0, stream>>>(s, d, (int)n);
  };
  for (int l = 0; l < NLAYERS; ++l) {
    conv(sa_w  + (size_t)l * 3 * DD + 2 * DD, wsa_v[l], DD);   // self-attn V proj
    conv(sa_ow + (size_t)l * DD,              wsa_o[l], DD);
    conv(ca_w  + (size_t)l * 3 * DD,          wca_q[l], DD);
    conv(ca_w  + (size_t)l * 3 * DD + DD,     wca_k[l], DD);
    conv(ca_w  + (size_t)l * 3 * DD + 2 * DD, wca_v[l], DD);
    conv(ca_ow + (size_t)l * DD,              wca_o[l], DD);
    conv(ff1_w + (size_t)l * DD,              wff1[l],  DD);
    conv(ff2_w + (size_t)l * DD,              wff2[l],  DD);
  }
  conv(glot_w, wglot, 2 * DD);

  // ---- K/V precompute: 4 big WMMA GEMMs ----
  for (int l = 0; l < NLAYERS; ++l) {
    kv_gemm_kernel<<<16384, 256, 0, stream>>>(memory, wca_k[l],
        ca_b + (size_t)l * 3 * D_MODEL + D_MODEL,     memK + (size_t)l * KVSZ);
    kv_gemm_kernel<<<16384, 256, 0, stream>>>(memory, wca_v[l],
        ca_b + (size_t)l * 3 * D_MODEL + 2 * D_MODEL, memV + (size_t)l * KVSZ);
  }

  // ---- x0 = go frame (zeros) ----
  zero_kernel<<<256, 256, 0, stream>>>(x, BATCH * D_MODEL);

  const size_t GATE_BASE = (size_t)BATCH * TSTEPS * 2 * D_MODEL;

  // ---- 48 sequential decoder steps; stream order carries the dependency chain ----
  for (int t = 0; t < TSTEPS; ++t) {
    for (int l = 0; l < NLAYERS; ++l) {
      const float* sab = sa_b + (size_t)l * 3 * D_MODEL;
      const float* cab = ca_b + (size_t)l * 3 * D_MODEL;
      // self-attn (len-1 target: softmax==1 -> V projection) + out_proj + LN1
      wmma_gemm_staged_kernel<<<16, 256, 0, stream>>>(x, wsa_v[l], sab + 2 * D_MODEL,
          nullptr, vbuf, D_MODEL, D_MODEL, D_MODEL, 0);
      wmma_gemm_staged_kernel<<<16, 256, 0, stream>>>(vbuf, wsa_o[l],
          sa_ob + (size_t)l * D_MODEL, x, ybuf, D_MODEL, D_MODEL, D_MODEL, 0);
      ln_kernel<<<64, 256, 0, stream>>>(ybuf, ln1_g + (size_t)l * D_MODEL,
          ln1_b + (size_t)l * D_MODEL, x);
      // cross-attention
      wmma_gemm_staged_kernel<<<16, 256, 0, stream>>>(x, wca_q[l], cab,
          nullptr, qbuf, D_MODEL, D_MODEL, D_MODEL, 0);
      attn_kernel<<<64, 512, 0, stream>>>(qbuf, memK + (size_t)l * KVSZ,
          memV + (size_t)l * KVSZ, obuf);
      wmma_gemm_staged_kernel<<<16, 256, 0, stream>>>(obuf, wca_o[l],
          ca_ob + (size_t)l * D_MODEL, x, ybuf, D_MODEL, D_MODEL, D_MODEL, 0);
      ln_kernel<<<64, 256, 0, stream>>>(ybuf, ln2_g + (size_t)l * D_MODEL,
          ln2_b + (size_t)l * D_MODEL, x);
      // FFN
      wmma_gemm_staged_kernel<<<16, 256, 0, stream>>>(x, wff1[l],
          ff1_b + (size_t)l * D_MODEL, nullptr, fbuf, D_MODEL, D_MODEL, D_MODEL, 1);
      wmma_gemm_staged_kernel<<<16, 256, 0, stream>>>(fbuf, wff2[l],
          ff2_b + (size_t)l * D_MODEL, x, ybuf, D_MODEL, D_MODEL, D_MODEL, 0);
      ln_kernel<<<64, 256, 0, stream>>>(ybuf, ln3_g + (size_t)l * D_MODEL,
          ln3_b + (size_t)l * D_MODEL, x);
    }
    // glottal: [B,2048] written strided into d_out[B,T,2D]
    wmma_gemm_staged_kernel<<<32, 256, 0, stream>>>(x, wglot, glot_b, nullptr,
        outf + (size_t)t * 2 * D_MODEL, (long)TSTEPS * 2 * D_MODEL,
        2 * D_MODEL, D_MODEL, 0);
    // gate: d_out[T,B,1]
    gate_kernel<<<64, 32, 0, stream>>>(x, gate_w, gate_b,
        outf + GATE_BASE + (size_t)t * BATCH);
  }
}
